// TransformerFusion_81449759801423
// MI455X (gfx1250) — matrix operand order repaired and checked by reading, not compiled, not run on hardware
//
#include <hip/hip_runtime.h>
#include <hip/hip_bf16.h>

typedef __bf16 bf16;
typedef __attribute__((ext_vector_type(16))) __bf16 v16bf;
typedef __attribute__((ext_vector_type(8)))  float  v8f;

// ---------------- problem dims ----------------
#define BB   16
#define DD   512
#define HH   48
#define WW   64
#define HWT  (HH * WW)          // 3072 tokens per image
#define MTOK (BB * HWT)         // 49152 rows (twins)
#define NHD  8
#define HDD  64
#define ZROWS 16384             // 1024 zones * 16 tokens
#define SCALE_ATTN 0.125f       // 64^-0.5

__device__ inline v8f vzero8() { v8f z = {0.f,0.f,0.f,0.f,0.f,0.f,0.f,0.f}; return z; }

// ---- CDNA5 async global->LDS copy (16B per lane), ASYNCcnt tracked ----
__device__ __attribute__((always_inline)) inline void async_cp16(void* lds, const void* gmem) {
    unsigned l = (unsigned)(size_t)lds;                     // low 32 bits of generic ptr = LDS offset
    unsigned long long g = (unsigned long long)(size_t)gmem;
    asm volatile("global_load_async_to_lds_b128 %0, %1, off" :: "v"(l), "v"(g) : "memory");
}
__device__ __attribute__((always_inline)) inline void wait_async0() {
    asm volatile("s_wait_asynccnt 0x0" ::: "memory");
}
__device__ __attribute__((always_inline)) inline void wait_async4() {
    asm volatile("s_wait_asynccnt 0x4" ::: "memory");
}

// ---- WMMA fragment loaders (bf16, LDS row-major tiles) ----
// A 16x32: lane L -> M=L&15, g=L>>4 ; halfs 0..7 = K(k0+g*8..+7), halfs 8..15 = K(k0+16+g*8..+7)
__device__ __attribute__((always_inline)) inline v16bf load_a_frag(const bf16* lds, int row0, int ld, int k0, int lane) {
    int m = lane & 15, g = lane >> 4;
    const bf16* p = lds + (size_t)(row0 + m) * ld + k0 + g * 8;
    union { v16bf v; uint4 u[2]; } r;
    r.u[0] = *(const uint4*)p;
    r.u[1] = *(const uint4*)(p + 16);
    return r.v;
}
// B 32x16: lane L -> N=L&15, g=L>>4 ; halfs 0..7 = K(k0+g*8..+7), halfs 8..15 = K(k0+16+g*8..+7). Tile stored [N][K].
__device__ __attribute__((always_inline)) inline v16bf load_b_frag(const bf16* lds, int n0, int ld, int k0, int lane) {
    int n = lane & 15, g = lane >> 4;
    const bf16* p = lds + (size_t)(n0 + n) * ld + k0 + g * 8;
    union { v16bf v; uint4 u[2]; } r;
    r.u[0] = *(const uint4*)p;
    r.u[1] = *(const uint4*)(p + 16);
    return r.v;
}

__device__ inline float gelu_tanh(float u) {
    float c = 0.7978845608028654f * (u + 0.044715f * u * u * u);
    return 0.5f * u * (1.0f + tanhf(c));
}

// ================= generic WMMA GEMM (templated epilogue) =================
// C[M,N] = act(A[M,K](bf16, row stride lda) * BwT[N,K](bf16) + bias) + resid
#define GBM 128
#define GBN 128
#define GBK 32
template <int ACT, bool HAS_BIAS, bool HAS_RESID, bool OUT_BF16>
__global__ __launch_bounds__(256) void gemm_bf16_kernel(
    const bf16* __restrict__ A, const bf16* __restrict__ BwT,
    const float* __restrict__ bias, const float* __restrict__ resid,
    void* __restrict__ Cout, int M, int N, int K, int lda)
{
    __shared__ __align__(16) bf16 as[2][GBM * GBK];
    __shared__ __align__(16) bf16 bs[2][GBN * GBK];
    const int tid = threadIdx.x, lane = tid & 31, wave = tid >> 5;
    const int wy = wave >> 2, wx = wave & 3;                 // 2 x 4 wave grid
    const int m0 = blockIdx.y * GBM, n0 = blockIdx.x * GBN;

    // per-thread staging coordinates: 2 chunks for A, 2 for B (16B each)
    const int r0 = tid >> 1, c0 = (tid & 1) * 16;            // 128 rows x {0,16}
    v8f acc[4][2];
#pragma unroll
    for (int mi = 0; mi < 4; ++mi)
#pragma unroll
        for (int ni = 0; ni < 2; ++ni) acc[mi][ni] = vzero8();

    // async-stage tile 0 into buffer 0 (each thread: 2x A-chunk, 2x B-chunk = 4 asyncs)
    {
        async_cp16(&as[0][r0 * GBK + c0],      &A[(size_t)(m0 + r0) * lda + c0]);
        async_cp16(&as[0][r0 * GBK + c0 + 8],  &A[(size_t)(m0 + r0) * lda + c0 + 8]);
        async_cp16(&bs[0][r0 * GBK + c0],      &BwT[(size_t)(n0 + r0) * K + c0]);
        async_cp16(&bs[0][r0 * GBK + c0 + 8],  &BwT[(size_t)(n0 + r0) * K + c0 + 8]);
    }
    int buf = 0;
    for (int k0 = 0; k0 < K; k0 += GBK) {
        if (k0 + GBK < K) {
            int nb2 = buf ^ 1, kn = k0 + GBK;
            async_cp16(&as[nb2][r0 * GBK + c0],     &A[(size_t)(m0 + r0) * lda + kn + c0]);
            async_cp16(&as[nb2][r0 * GBK + c0 + 8], &A[(size_t)(m0 + r0) * lda + kn + c0 + 8]);
            async_cp16(&bs[nb2][r0 * GBK + c0],     &BwT[(size_t)(n0 + r0) * K + kn + c0]);
            async_cp16(&bs[nb2][r0 * GBK + c0 + 8], &BwT[(size_t)(n0 + r0) * K + kn + c0 + 8]);
            wait_async4();   // async ops retire in order: <=4 outstanding => current tile landed
        } else {
            wait_async0();
        }
        __syncthreads();
#pragma unroll
        for (int mi = 0; mi < 4; ++mi) {
            v16bf afr = load_a_frag(as[buf], wy * 64 + mi * 16, GBK, 0, lane);
#pragma unroll
            for (int ni = 0; ni < 2; ++ni) {
                v16bf bfr = load_b_frag(bs[buf], wx * 32 + ni * 16, GBK, 0, lane);
                acc[mi][ni] = __builtin_amdgcn_wmma_f32_16x16x32_bf16(
                    false, afr, false, bfr, (short)0, acc[mi][ni], false, false);
            }
        }
        __syncthreads();    // safe to overwrite this buffer next iteration
        buf ^= 1;
    }
    // epilogue (C layout: n = lane&15, m = i + 8*(lane>>4)) -- branch-free per instantiation
    const int nb = lane & 15, g = lane >> 4;
#pragma unroll
    for (int mi = 0; mi < 4; ++mi)
#pragma unroll
        for (int ni = 0; ni < 2; ++ni)
#pragma unroll
            for (int i = 0; i < 8; ++i) {
                int m = m0 + wy * 64 + mi * 16 + i + 8 * g;
                int n = n0 + wx * 32 + ni * 16 + nb;
                float v = acc[mi][ni][i];
                if (HAS_BIAS) v += bias[n];
                if (ACT == 1) v = gelu_tanh(v);
                else if (ACT == 2) v = fmaxf(v, 0.f);
                if (HAS_RESID) v += resid[(size_t)m * N + n];
                if (OUT_BF16) ((bf16*)Cout)[(size_t)m * N + n] = (bf16)v;
                else          ((float*)Cout)[(size_t)m * N + n] = v;
            }
}

// ================= Twins windowed attention =================
// one wave per (window, head); qkv: bf16 [MTOK, 1536], o: bf16 [MTOK, 512]
__global__ __launch_bounds__(32) void twins_attn_kernel(
    const bf16* __restrict__ qkv, bf16* __restrict__ o)
{
    __shared__ __align__(16) bf16 qs[64 * 64];
    __shared__ __align__(16) bf16 ks[64 * 64];
    __shared__ __align__(16) bf16 vs[64 * 64];   // transposed: [d][tok]
    __shared__ __align__(16) bf16 ps[16 * 64];
    const int lane = threadIdx.x;
    const int head = blockIdx.x & 7;
    const int win  = blockIdx.x >> 3;           // 0..767
    const int b  = win / 48;
    const int r  = win % 48;
    const int bh = r >> 3, bw = r & 7;

    // stage: Q,K via async DMA (contiguous 16B chunks); V transposed via VGPRs
    for (int ch = lane; ch < 512; ch += 32) {
        int tok = ch >> 3, c8 = (ch & 7) * 8;
        int ih = tok >> 3, iw = tok & 7;
        size_t row = (size_t)b * HWT + (size_t)(bh * 8 + ih) * WW + (bw * 8 + iw);
        const bf16* base = qkv + row * 1536 + head * 64;
        async_cp16(&qs[tok * 64 + c8], base + c8);
        async_cp16(&ks[tok * 64 + c8], base + 512 + c8);
        union { uint4 u; bf16 h[8]; } t;
        t.u = *(const uint4*)(base + 1024 + c8);
#pragma unroll
        for (int j = 0; j < 8; ++j) vs[(c8 + j) * 64 + tok] = t.h[j];
    }
    wait_async0();
    __syncthreads();

    const int nb = lane & 15, g = lane >> 4;
    for (int strip = 0; strip < 4; ++strip) {
        v8f acc[4];
#pragma unroll
        for (int nt = 0; nt < 4; ++nt) acc[nt] = vzero8();
#pragma unroll
        for (int kk = 0; kk < 64; kk += 32) {
            v16bf a = load_a_frag(qs, strip * 16, 64, kk, lane);
#pragma unroll
            for (int nt = 0; nt < 4; ++nt)
                acc[nt] = __builtin_amdgcn_wmma_f32_16x16x32_bf16(
                    false, a, false, load_b_frag(ks, nt * 16, 64, kk, lane),
                    (short)0, acc[nt], false, false);
        }
        // row softmax (row = i + 8*g lives across 16 lanes of a half-wave)
#pragma unroll
        for (int i = 0; i < 8; ++i) {
            float m = -1e30f;
            float e[4];
#pragma unroll
            for (int nt = 0; nt < 4; ++nt) { float s = acc[nt][i] * SCALE_ATTN; acc[nt][i] = s; m = fmaxf(m, s); }
#pragma unroll
            for (int off = 1; off < 16; off <<= 1) m = fmaxf(m, __shfl_xor(m, off, 32));
            float s = 0.f;
#pragma unroll
            for (int nt = 0; nt < 4; ++nt) { e[nt] = __expf(acc[nt][i] - m); s += e[nt]; }
#pragma unroll
            for (int off = 1; off < 16; off <<= 1) s += __shfl_xor(s, off, 32);
            float inv = 1.0f / s;
            int prow = i + 8 * g;
#pragma unroll
            for (int nt = 0; nt < 4; ++nt) ps[prow * 64 + nt * 16 + nb] = (bf16)(e[nt] * inv);
        }
        __syncthreads();
        // O = P @ V
        v8f oac[4];
#pragma unroll
        for (int nt = 0; nt < 4; ++nt) oac[nt] = vzero8();
#pragma unroll
        for (int kk = 0; kk < 64; kk += 32) {
            v16bf a = load_a_frag(ps, 0, 64, kk, lane);
#pragma unroll
            for (int nt = 0; nt < 4; ++nt)
                oac[nt] = __builtin_amdgcn_wmma_f32_16x16x32_bf16(
                    false, a, false, load_b_frag(vs, nt * 16, 64, kk, lane),
                    (short)0, oac[nt], false, false);
        }
#pragma unroll
        for (int nt = 0; nt < 4; ++nt)
#pragma unroll
            for (int i = 0; i < 8; ++i) {
                int tok = strip * 16 + i + 8 * g;
                int d = nt * 16 + nb;
                int ih = tok >> 3, iw = tok & 7;
                size_t row = (size_t)b * HWT + (size_t)(bh * 8 + ih) * WW + (bw * 8 + iw);
                o[row * 512 + head * 64 + d] = (bf16)oac[nt][i];
            }
        __syncthreads();
    }
}

// ================= small kernels =================
// weight convert + transpose: src f32 [K,N] -> dst bf16 [N,K]
__global__ void w_transpose_bf16_kernel(const float* __restrict__ s, bf16* __restrict__ d,
                                        int K, int N) {
    int i = blockIdx.x * 256 + threadIdx.x;
    if (i < K * N) {
        int k = i / N, n = i % N;
        d[(size_t)n * K + k] = (bf16)s[i];
    }
}

__global__ void emb_kernel(const float* __restrict__ x, const float* __restrict__ pos,
                           float* __restrict__ emb) {
    size_t i = (size_t)blockIdx.x * 256 + threadIdx.x;     // over MTOK*DD
    size_t d = i & 511;
    size_t bt = i >> 9;
    size_t t = bt % HWT, b = bt / HWT;
    emb[i] = x[(b * DD + d) * HWT + t] + pos[t * DD + d];
}

// layernorm over 512, bf16 out with output stride
__global__ __launch_bounds__(256) void ln512_kernel(const float* __restrict__ in,
    const float* __restrict__ gg, const float* __restrict__ bbv,
    bf16* __restrict__ out, int ld_out, int M)
{
    int lane = threadIdx.x & 31, wave = threadIdx.x >> 5;
    int row = blockIdx.x * 8 + wave;
    if (row >= M) return;
    const float* p = in + (size_t)row * 512;
    float vals[16]; float s = 0.f;
#pragma unroll
    for (int j = 0; j < 16; ++j) { vals[j] = p[lane + j * 32]; s += vals[j]; }
#pragma unroll
    for (int off = 1; off < 32; off <<= 1) s += __shfl_xor(s, off, 32);
    float mean = s * (1.0f / 512.0f);
    float vsum = 0.f;
#pragma unroll
    for (int j = 0; j < 16; ++j) { float d = vals[j] - mean; vsum += d * d; }
#pragma unroll
    for (int off = 1; off < 32; off <<= 1) vsum += __shfl_xor(vsum, off, 32);
    float inv = rsqrtf(vsum * (1.0f / 512.0f) + 1e-5f);
    bf16* q = out + (size_t)row * ld_out;
#pragma unroll
    for (int j = 0; j < 16; ++j) {
        int c = lane + j * 32;
        q[c] = (bf16)((vals[j] - mean) * inv * gg[c] + bbv[c]);
    }
}

// zf = (zone + LN(m_in)) * mask, f32 out
__global__ __launch_bounds__(256) void ln_resid_mask_kernel(const float* __restrict__ in,
    const float* __restrict__ gg, const float* __restrict__ bbv,
    const float* __restrict__ zone, const unsigned char* __restrict__ mask,
    float* __restrict__ zf, int M)
{
    int lane = threadIdx.x & 31, wave = threadIdx.x >> 5;
    int row = blockIdx.x * 8 + wave;
    if (row >= M) return;
    const float* p = in + (size_t)row * 512;
    float vals[16]; float s = 0.f;
#pragma unroll
    for (int j = 0; j < 16; ++j) { vals[j] = p[lane + j * 32]; s += vals[j]; }
#pragma unroll
    for (int off = 1; off < 32; off <<= 1) s += __shfl_xor(s, off, 32);
    float mean = s * (1.0f / 512.0f);
    float vsum = 0.f;
#pragma unroll
    for (int j = 0; j < 16; ++j) { float d = vals[j] - mean; vsum += d * d; }
#pragma unroll
    for (int off = 1; off < 32; off <<= 1) vsum += __shfl_xor(vsum, off, 32);
    float inv = rsqrtf(vsum * (1.0f / 512.0f) + 1e-5f);
    float mk = mask[row >> 4] ? 1.0f : 0.0f;
    float* q = zf + (size_t)row * 512;
    const float* zp = zone + (size_t)row * 512;
#pragma unroll
    for (int j = 0; j < 16; ++j) {
        int c = lane + j * 32;
        float ln = (vals[j] - mean) * inv * gg[c] + bbv[c];
        q[c] = (zp[c] + ln) * mk;
    }
}

// zone extraction: emb -> zone f32 + cat(bf16, left half, ld 1024)
__global__ void zone_extract_kernel(const float* __restrict__ emb,
                                    float* __restrict__ zone, bf16* __restrict__ cat) {
    size_t i = (size_t)blockIdx.x * 256 + threadIdx.x;     // over ZROWS*512
    int d = (int)(i & 511);
    int row = (int)(i >> 9);
    int l = row & 15, n = row >> 4;
    int z = n & 63, b = n >> 6;
    int zy = z >> 3, zx = z & 7, py = l >> 2, px = l & 3;
    int y = 8 + zy * 4 + py, xq = 16 + zx * 4 + px;
    float v = emb[((size_t)b * HWT + y * WW + xq) * 512 + d];
    zone[(size_t)row * 512 + d] = v;
    cat[(size_t)row * 1024 + d] = (bf16)v;
}

__global__ void f1_build_kernel(const float* __restrict__ feat1, const float* __restrict__ pos2,
                                bf16* __restrict__ f1) {
    size_t i = (size_t)blockIdx.x * 256 + threadIdx.x;     // over ZROWS*512
    int d = (int)(i & 511);
    int s = (int)((i >> 9) & 15);
    f1[i] = (bf16)(feat1[i] + pos2[s * 512 + d]);
}

// LoFTR linear-attention core: per (zone, head) block
__global__ __launch_bounds__(128) void loftr_core_kernel(
    const float* __restrict__ qf, const float* __restrict__ kf, const float* __restrict__ vf,
    bf16* __restrict__ msg)
{
    __shared__ float Q[16 * 64], Kk[16 * 64], V[16 * 64];
    __shared__ float KV[64 * 64], Ksum[64], Z[16];
    const int tid = threadIdx.x;
    const int h = blockIdx.x & 7;
    const int n = blockIdx.x >> 3;
    for (int idx = tid; idx < 1024; idx += 128) {
        int s = idx >> 6, d = idx & 63;
        size_t row = (size_t)n * 16 + s;
        float q = qf[row * 512 + h * 64 + d];
        float k = kf[row * 512 + h * 64 + d];
        Q[idx]  = q > 0.f ? q + 1.0f : __expf(q);
        Kk[idx] = k > 0.f ? k + 1.0f : __expf(k);
        V[idx]  = vf[row * 512 + h * 64 + d];
    }
    __syncthreads();
    if (tid < 64) {
        float s = 0.f;
        for (int ss = 0; ss < 16; ++ss) s += Kk[ss * 64 + tid];
        Ksum[tid] = s;
    }
    for (int idx = tid; idx < 4096; idx += 128) {
        int d = idx >> 6, e = idx & 63;
        float a = 0.f;
        for (int s = 0; s < 16; ++s) a += Kk[s * 64 + d] * V[s * 64 + e];
        KV[idx] = a;
    }
    __syncthreads();
    if (tid < 16) {
        float z = 0.f;
        for (int d = 0; d < 64; ++d) z += Q[tid * 64 + d] * Ksum[d];
        Z[tid] = 1.0f / (z + 1e-6f);
    }
    __syncthreads();
    for (int idx = tid; idx < 1024; idx += 128) {
        int l = idx >> 6, e = idx & 63;
        float a = 0.f;
        for (int d = 0; d < 64; ++d) a += Q[l * 64 + d] * KV[d * 64 + e];
        msg[((size_t)n * 16 + l) * 512 + h * 64 + e] = (bf16)(a * Z[l]);
    }
}

__global__ void scatter_add_kernel(const float* __restrict__ zf, float* __restrict__ feat0) {
    size_t i = (size_t)blockIdx.x * 256 + threadIdx.x;     // over ZROWS*512
    int d = (int)(i & 511);
    int row = (int)(i >> 9);
    int l = row & 15, n = row >> 4;
    int z = n & 63, b = n >> 6;
    int zy = z >> 3, zx = z & 7, py = l >> 2, px = l & 3;
    int y = 8 + zy * 4 + py, xq = 16 + zx * 4 + px;
    feat0[((size_t)b * HWT + y * WW + xq) * 512 + d] += zf[i];
}

__global__ void final_out_kernel(const float* __restrict__ feat0, float* __restrict__ out) {
    size_t i = (size_t)blockIdx.x * 256 + threadIdx.x;     // over B*D*H*W (output linear)
    size_t t = i % HWT;
    size_t bd = i / HWT;
    size_t d = bd % DD, b = bd / DD;
    out[i] = feat0[(b * HWT + t) * DD + d];
}

// ================= launcher =================
template <int ACT, bool HB, bool HR, bool OB>
static inline void launch_gemm(const bf16* A, const bf16* BwT, const float* bias,
                               const float* resid, void* C, int M, int N, int K,
                               int lda, hipStream_t s) {
    dim3 g(N / GBN, M / GBM);
    gemm_bf16_kernel<ACT, HB, HR, OB><<<g, 256, 0, s>>>(A, BwT, bias, resid, C, M, N, K, lda);
}

extern "C" void kernel_launch(void* const* d_in, const int* in_sizes, int n_in,
                              void* d_out, int out_size, void* d_ws, size_t ws_size,
                              hipStream_t stream) {
    const float* x        = (const float*)d_in[0];
    const float* feat1    = (const float*)d_in[1];
    const unsigned char* mask = (const unsigned char*)d_in[2];
    const float* pos_enc  = (const float*)d_in[3];
    const float* pos_enc2 = (const float*)d_in[4];
    const float* tw_ln1_g = (const float*)d_in[5];
    const float* tw_ln1_b = (const float*)d_in[6];
    const float* tw_qkv_w = (const float*)d_in[7];
    const float* tw_qkv_b = (const float*)d_in[8];
    const float* tw_proj_w= (const float*)d_in[9];
    const float* tw_proj_b= (const float*)d_in[10];
    const float* tw_ln2_g = (const float*)d_in[11];
    const float* tw_ln2_b = (const float*)d_in[12];
    const float* tw_fc1_w = (const float*)d_in[13];
    const float* tw_fc1_b = (const float*)d_in[14];
    const float* tw_fc2_w = (const float*)d_in[15];
    const float* tw_fc2_b = (const float*)d_in[16];
    const float* lf_q_w   = (const float*)d_in[17];
    const float* lf_k_w   = (const float*)d_in[18];
    const float* lf_v_w   = (const float*)d_in[19];
    const float* lf_merge_w=(const float*)d_in[20];
    const float* lf_ln1_g = (const float*)d_in[21];
    const float* lf_ln1_b = (const float*)d_in[22];
    const float* lf_mlp1_w= (const float*)d_in[23];
    const float* lf_mlp2_w= (const float*)d_in[24];
    const float* lf_ln2_g = (const float*)d_in[25];
    const float* lf_ln2_b = (const float*)d_in[26];

    char* ws = (char*)d_ws;
    // ---- workspace layout (bytes) ----
    const size_t SZ_F32_BIG  = (size_t)MTOK * DD * 4;        // 100,663,296
    const size_t SZ_BF_BIG   = (size_t)MTOK * 2048 * 2;      // 201,326,592
    const size_t SZ_F32_Z    = (size_t)ZROWS * DD * 4;       //  33,554,432
    const size_t SZ_BF_Z     = (size_t)ZROWS * DD * 2;       //  16,777,216
    const size_t oA = 0;                                     // emb_tok, later feat0
    const size_t oB = oA + SZ_F32_BIG;                       // x1
    const size_t oC = oB + SZ_F32_BIG;                       // qkv/fc1 bf16, later loftr temps
    const size_t oD = oC + SZ_BF_BIG;                        // ln bf16 + o bf16, later zone/zf/msg
    const size_t oE = oD + SZ_F32_BIG;                       // cat bf16 [ZROWS,1024]
    const size_t oF = oE + (size_t)ZROWS * 1024 * 2;         // f1 bf16
    const size_t oG = oF + SZ_BF_Z;                          // weights bf16 (transposed [N][K])

    float* emb_tok = (float*)(ws + oA);
    float* feat0   = (float*)(ws + oA);
    float* x1      = (float*)(ws + oB);
    bf16*  qkv_bf  = (bf16*)(ws + oC);
    bf16*  fc1_out = (bf16*)(ws + oC);
    float* qf      = (float*)(ws + oC);
    float* kf      = (float*)(ws + oC + SZ_F32_Z);
    float* vf      = (float*)(ws + oC + 2 * SZ_F32_Z);
    float* merge_o = (float*)(ws + oC + 3 * SZ_F32_Z);
    bf16*  mlp1_o  = (bf16*)(ws + oC + 4 * SZ_F32_Z);
    float* m_out   = (float*)(ws + oC + 5 * SZ_F32_Z);
    bf16*  ln_a    = (bf16*)(ws + oD);
    bf16*  o_bf    = (bf16*)(ws + oD + (size_t)MTOK * DD * 2);
    float* zone_f  = (float*)(ws + oD);
    float* zf      = (float*)(ws + oD + SZ_F32_Z);
    bf16*  msg_bf  = (bf16*)(ws + oD + 2 * SZ_F32_Z);
    bf16*  cat     = (bf16*)(ws + oE);
    bf16*  f1      = (bf16*)(ws + oF);

    // transposed weight slots (bf16, [N][K])
    size_t wo = oG;
    bf16* w_qkv  = (bf16*)(ws + wo); wo += (size_t)512 * 1536 * 2;
    bf16* w_proj = (bf16*)(ws + wo); wo += (size_t)512 * 512 * 2;
    bf16* w_fc1  = (bf16*)(ws + wo); wo += (size_t)512 * 2048 * 2;
    bf16* w_fc2  = (bf16*)(ws + wo); wo += (size_t)2048 * 512 * 2;
    bf16* w_q    = (bf16*)(ws + wo); wo += (size_t)512 * 512 * 2;
    bf16* w_k    = (bf16*)(ws + wo); wo += (size_t)512 * 512 * 2;
    bf16* w_v    = (bf16*)(ws + wo); wo += (size_t)512 * 512 * 2;
    bf16* w_mrg  = (bf16*)(ws + wo); wo += (size_t)512 * 512 * 2;
    bf16* w_m1   = (bf16*)(ws + wo); wo += (size_t)1024 * 1024 * 2;
    bf16* w_m2   = (bf16*)(ws + wo); wo += (size_t)1024 * 512 * 2;

    auto cvt = [&](const float* s, bf16* d, int K, int N) {
        int n = K * N;
        w_transpose_bf16_kernel<<<(n + 255) / 256, 256, 0, stream>>>(s, d, K, N);
    };
    cvt(tw_qkv_w,  w_qkv,  512, 1536);
    cvt(tw_proj_w, w_proj, 512, 512);
    cvt(tw_fc1_w,  w_fc1,  512, 2048);
    cvt(tw_fc2_w,  w_fc2,  2048, 512);
    cvt(lf_q_w,    w_q,    512, 512);
    cvt(lf_k_w,    w_k,    512, 512);
    cvt(lf_v_w,    w_v,    512, 512);
    cvt(lf_merge_w,w_mrg,  512, 512);
    cvt(lf_mlp1_w, w_m1,   1024, 1024);
    cvt(lf_mlp2_w, w_m2,   1024, 512);

    // ---- Twins block ----
    emb_kernel<<<(MTOK * DD) / 256, 256, 0, stream>>>(x, pos_enc, emb_tok);
    ln512_kernel<<<MTOK / 8, 256, 0, stream>>>(emb_tok, tw_ln1_g, tw_ln1_b, ln_a, 512, MTOK);
    launch_gemm<0, true, false, true>(ln_a, w_qkv, tw_qkv_b, nullptr, qkv_bf,
                                      MTOK, 1536, 512, 512, stream);
    twins_attn_kernel<<<768 * NHD, 32, 0, stream>>>(qkv_bf, o_bf);
    launch_gemm<0, true, true, false>(o_bf, w_proj, tw_proj_b, emb_tok, x1,
                                      MTOK, 512, 512, 512, stream);
    ln512_kernel<<<MTOK / 8, 256, 0, stream>>>(x1, tw_ln2_g, tw_ln2_b, ln_a, 512, MTOK);
    launch_gemm<1, true, false, true>(ln_a, w_fc1, tw_fc1_b, nullptr, fc1_out,
                                      MTOK, 2048, 512, 512, stream);
    // zone extraction must happen while emb_tok is intact (before fc2 overwrites region A)
    zone_extract_kernel<<<(ZROWS * DD) / 256, 256, 0, stream>>>(emb_tok, zone_f, cat);
    launch_gemm<0, true, true, false>(fc1_out, w_fc2, tw_fc2_b, x1, feat0,
                                      MTOK, 512, 2048, 2048, stream);

    // ---- LoFTR block ----
    f1_build_kernel<<<(ZROWS * DD) / 256, 256, 0, stream>>>(feat1, pos_enc2, f1);
    launch_gemm<0, false, false, false>(cat, w_q, nullptr, nullptr, qf,
                                        ZROWS, 512, 512, 1024, stream);
    launch_gemm<0, false, false, false>(f1, w_k, nullptr, nullptr, kf,
                                        ZROWS, 512, 512, 512, stream);
    launch_gemm<0, false, false, false>(f1, w_v, nullptr, nullptr, vf,
                                        ZROWS, 512, 512, 512, stream);
    loftr_core_kernel<<<1024 * NHD, 128, 0, stream>>>(qf, kf, vf, msg_bf);
    launch_gemm<0, false, false, false>(msg_bf, w_mrg, nullptr, nullptr, merge_o,
                                        ZROWS, 512, 512, 512, stream);
    ln512_kernel<<<ZROWS / 8, 256, 0, stream>>>(merge_o, lf_ln1_g, lf_ln1_b, cat + 512, 1024, ZROWS);
    launch_gemm<2, false, false, true>(cat, w_m1, nullptr, nullptr, mlp1_o,
                                       ZROWS, 1024, 1024, 1024, stream);
    launch_gemm<0, false, false, false>(mlp1_o, w_m2, nullptr, nullptr, m_out,
                                        ZROWS, 512, 1024, 1024, stream);
    ln_resid_mask_kernel<<<ZROWS / 8, 256, 0, stream>>>(m_out, lf_ln2_g, lf_ln2_b,
                                                        zone_f, mask, zf, ZROWS);
    scatter_add_kernel<<<(ZROWS * DD) / 256, 256, 0, stream>>>(zf, feat0);
    final_out_kernel<<<(MTOK * DD) / 256, 256, 0, stream>>>(feat0, (float*)d_out);
}
